// TimeAwareGRUCell_49709951484728
// MI455X (gfx1250) — matrix operand order repaired and checked by reading, not compiled, not run on hardware
//
#include <hip/hip_runtime.h>
#include <hip/hip_bf16.h>
#include <math.h>

#define B_ROWS 16384
#define D_IN   512
#define D_H    1024
#define K_CAT  (D_H + D_IN)   // 1536

#define MT 128          // block tile M
#define NT 128          // block tile N
#define KS 64           // K step per LDS stage
#define LDSTR 72        // padded LDS row stride (bf16 elems)

typedef __attribute__((ext_vector_type(16))) __bf16 v16bf;
typedef __attribute__((ext_vector_type(8)))  __bf16 v8bf;
typedef __attribute__((ext_vector_type(8)))  float  v8f;

__device__ __forceinline__ float sigmoidf_(float x) {
  return 1.0f / (1.0f + __expf(-x));
}

__device__ __forceinline__ v16bf cat16(v8bf lo, v8bf hi) {
  return __builtin_shufflevector(lo, hi,
      0, 1, 2, 3, 4, 5, 6, 7, 8, 9, 10, 11, 12, 13, 14, 15);
}

__device__ __forceinline__ void async_tile_b128(unsigned lds_byte_off,
                                                const __bf16* gaddr) {
  // GLOBAL_LOAD_ASYNC_TO_LDS_B128: VDST = per-lane LDS byte address,
  // VADDR = 64-bit global address. Tracked by ASYNCcnt.
  asm volatile("global_load_async_to_lds_b128 %0, %1, off"
               :: "v"(lds_byte_off), "v"(gaddr)
               : "memory");
}

__device__ __forceinline__ void wait_asynccnt0() {
#if __has_builtin(__builtin_amdgcn_s_wait_asynccnt)
  __builtin_amdgcn_s_wait_asynccnt(0);
#else
  asm volatile("s_wait_asynccnt 0x0" ::: "memory");
#endif
}

// ---------------- pack / convert kernels (8-wide vectorized) ----------------

__device__ __forceinline__ v8bf cvt8(const float* __restrict__ src) {
  v8bf r;
#pragma unroll
  for (int i = 0; i < 8; ++i) r[i] = (__bf16)src[i];
  return r;
}

__global__ void pack_hx_kernel(const float* __restrict__ h,
                               const float* __restrict__ x,
                               __bf16* __restrict__ hx, int total8) {
  int i = blockIdx.x * blockDim.x + threadIdx.x;
  if (i >= total8) return;
  const int kc8 = K_CAT / 8;            // 192
  int m = i / kc8, k = (i - m * kc8) * 8;
  const float* src = (k < D_H) ? (h + (size_t)m * D_H + k)
                               : (x + (size_t)m * D_IN + (k - D_H));
  *(v8bf*)(hx + (size_t)m * K_CAT + k) = cvt8(src);
}

struct PackW1Args {
  const float* Wh[4];
  const float* Wx[4];
  __bf16* out;
};

__global__ void pack_w1_kernel(PackW1Args p, int total8) {
  int i = blockIdx.x * blockDim.x + threadIdx.x;
  if (i >= total8) return;
  const int kc8 = K_CAT / 8;
  int n = i / kc8, k = (i - n * kc8) * 8;
  int c = n >> 10, nn = n & (D_H - 1);
  const float* src = (k < D_H) ? (p.Wh[c] + (size_t)nn * D_H + k)
                               : (p.Wx[c] + (size_t)nn * D_IN + (k - D_H));
  *(v8bf*)(p.out + (size_t)n * K_CAT + k) = cvt8(src);
}

__global__ void cvt_bf16_kernel(const float* __restrict__ in,
                                __bf16* __restrict__ out, int total8) {
  int i = blockIdx.x * blockDim.x + threadIdx.x;
  if (i >= total8) return;
  *(v8bf*)(out + (size_t)i * 8) = cvt8(in + (size_t)i * 8);
}

__global__ void pack_w3_kernel(const float* __restrict__ Wh,
                               const float* __restrict__ Wx,
                               __bf16* __restrict__ out, int total8) {
  int i = blockIdx.x * blockDim.x + threadIdx.x;
  if (i >= total8) return;
  const int kc8 = K_CAT / 8;
  int n = i / kc8, k = (i - n * kc8) * 8;
  const float* src = (k < D_H) ? (Wh + (size_t)n * D_H + k)
                               : (Wx + (size_t)n * D_IN + (k - D_H));
  *(v8bf*)(out + (size_t)n * K_CAT + k) = cvt8(src);
}

__global__ void u_write_kernel(const float* __restrict__ r,
                               const float* __restrict__ h,
                               __bf16* __restrict__ hx, int total8) {
  int i = blockIdx.x * blockDim.x + threadIdx.x;
  if (i >= total8) return;
  const int dh8 = D_H / 8;
  int m = i / dh8, k = (i - m * dh8) * 8;
  const float* rp = r + (size_t)m * D_H + k;
  const float* hp = h + (size_t)m * D_H + k;
  v8bf u;
#pragma unroll
  for (int e = 0; e < 8; ++e) u[e] = (__bf16)(rp[e] * hp[e]);
  *(v8bf*)(hx + (size_t)m * K_CAT + k) = u;
}

// ------- async double-buffered WMMA GEMM (bf16 -> f32), A @ W^T -------
// 256 threads = 8 waves (4 along M x 2 along N). Block tile 128x128.
// Wave tile 32(M) x 64(N) = 2x4 V_WMMA_F32_16X16X32_BF16 accumulators.
// Tiles staged via GLOBAL_LOAD_ASYNC_TO_LDS_B128 (ASYNCcnt), double buffered.

struct GemmParams {
  const __bf16* A; const __bf16* W;
  int K;
  // mode 1 epilogue:
  const float* delta; const float* Wst;
  const float* bs; const float* bT; const float* br; const float* bz;
  __bf16* s_bf; float* Tpart; float* rbuf; float* zbuf;
  // mode 3 epilogue:
  const float* b3; const float* hprev; float* out;
};

#define A_ELEMS (MT * LDSTR)            // elems per A buffer
#define B_BASE  (2 * MT * LDSTR)        // elem offset of B buffers
#define B_ELEMS (NT * LDSTR)

template <int MODE>
__global__ __launch_bounds__(256)
void gemm_wmma_kernel(GemmParams p) {
  // single LDS object => workgroup LDS byte offset 0 for manual addressing
  __shared__ __bf16 smem[2 * MT * LDSTR + 2 * NT * LDSTR];

  const int K    = p.K;
  const int t    = threadIdx.x;
  const int lane = t & 31;
  const int wave = t >> 5;
  const int l15  = lane & 15;
  const int lhi  = lane >> 4;

  const int rowB = blockIdx.y * MT;
  const int colB = blockIdx.x * NT;
  const int wm   = (wave & 3) * 32;     // wave M offset inside block tile
  const int wn   = (wave >> 2) * 64;    // wave N offset inside block tile

  // cooperative staging: each thread DMAs 64B of A and 64B of W per tile
  const int ldRow = t >> 1;             // 0..127
  const int ldCol = (t & 1) * 32;       // elem offset 0 or 32

  const __bf16* __restrict__ Ag = p.A + (size_t)(rowB + ldRow) * K + ldCol;
  const __bf16* __restrict__ Wg = p.W + (size_t)(colB + ldRow) * K + ldCol;

  const unsigned aOffBase = (unsigned)((ldRow * LDSTR + ldCol) * 2);
  const unsigned bOffBase = (unsigned)((B_BASE + ldRow * LDSTR + ldCol) * 2);

  v8f acc[2][4] = {};

  const int nIter = K / KS;

  auto stage_async = [&](int gk, int b) {
    const unsigned ao = aOffBase + (unsigned)(b * A_ELEMS * 2);
    const unsigned bo = bOffBase + (unsigned)(b * B_ELEMS * 2);
#pragma unroll
    for (int i = 0; i < 4; ++i) {
      async_tile_b128(ao + i * 16, Ag + gk + i * 8);
      async_tile_b128(bo + i * 16, Wg + gk + i * 8);
    }
  };

  // prologue: DMA tile 0 into buffer 0
  stage_async(0, 0);
  wait_asynccnt0();
  __syncthreads();

  for (int it = 0; it < nIter; ++it) {
    const int buf = it & 1;
    const bool more = (it + 1) < nIter;

    // kick off DMA of the NEXT tile into the other buffer (overlaps WMMAs)
    if (more) stage_async((it + 1) * KS, 1 - buf);

    const __bf16* aB = smem + buf * A_ELEMS;
    const __bf16* bB = smem + B_BASE + buf * B_ELEMS;

#pragma unroll
    for (int kb = 0; kb < KS; kb += 32) {
#pragma unroll
      for (int ms = 0; ms < 2; ++ms) {
        const __bf16* ar = aB + (wm + ms * 16 + l15) * LDSTR + kb + lhi * 8;
        v8bf alo = *(const v8bf*)(ar);
        v8bf ahi = *(const v8bf*)(ar + 16);
        v16bf afrag = cat16(alo, ahi);
#pragma unroll
        for (int j = 0; j < 4; ++j) {
          const __bf16* br = bB + (wn + j * 16 + l15) * LDSTR + kb + lhi * 8;
          v8bf blo = *(const v8bf*)(br);
          v8bf bhi = *(const v8bf*)(br + 16);
          acc[ms][j] = __builtin_amdgcn_wmma_f32_16x16x32_bf16(
              false, afrag, false, cat16(blo, bhi), (short)0, acc[ms][j],
              false, false);
        }
      }
    }

    if (more) {
      wait_asynccnt0();   // this wave's DMA into buf^1 complete
      __syncthreads();    // all waves' DMA complete + reads of buf done
    }
  }

  // Epilogue. C/D layout: lane -> N = lane&15, VGPR v -> M = v + (lane>>4)*8
#pragma unroll
  for (int ms = 0; ms < 2; ++ms) {
    const int mbase = rowB + wm + ms * 16 + lhi * 8;
#pragma unroll
    for (int j = 0; j < 4; ++j) {
      const int n = colB + wn + j * 16 + l15;
#pragma unroll
      for (int v = 0; v < 8; ++v) {
        const int m = mbase + v;
        float a = acc[ms][j][v];
        if (MODE == 1) {
          const int c = n >> 10, nn = n & (D_H - 1);
          const size_t idx = (size_t)m * D_H + nn;
          if (c == 0) {
            p.s_bf[idx] = (__bf16)tanhf(a + p.delta[m] * p.Wst[nn] + p.bs[nn]);
          } else if (c == 1) {
            p.Tpart[idx] = a + p.bT[nn];
          } else if (c == 2) {
            p.rbuf[idx] = sigmoidf_(a + p.br[nn]);
          } else {
            p.zbuf[idx] = sigmoidf_(a + p.bz[nn]);
          }
        } else if (MODE == 2) {
          const size_t idx = (size_t)m * D_H + n;
          p.Tpart[idx] = sigmoidf_(a + p.Tpart[idx]);
        } else {
          const size_t idx = (size_t)m * D_H + n;
          const float ht = tanhf(a + p.b3[n]);
          const float z = p.zbuf[idx];
          const float T = p.Tpart[idx];
          const float h = p.hprev[idx];
          p.out[idx] = (1.0f - z) * (T * h) + z * ht;
        }
      }
    }
  }
}

// ---------------- launch ----------------

extern "C" void kernel_launch(void* const* d_in, const int* in_sizes, int n_in,
                              void* d_out, int out_size, void* d_ws, size_t ws_size,
                              hipStream_t stream) {
  const float* x     = (const float*)d_in[0];
  const float* delta = (const float*)d_in[1];
  const float* h     = (const float*)d_in[2];
  const float* W_sh  = (const float*)d_in[3];
  const float* W_sx  = (const float*)d_in[4];
  const float* W_st  = (const float*)d_in[5];
  const float* b_s   = (const float*)d_in[6];
  const float* WTh   = (const float*)d_in[7];
  const float* WTx   = (const float*)d_in[8];
  const float* WTs   = (const float*)d_in[9];
  const float* b_T   = (const float*)d_in[10];
  const float* W_rh  = (const float*)d_in[11];
  const float* W_rx  = (const float*)d_in[12];
  const float* b_r   = (const float*)d_in[13];
  const float* W_zh  = (const float*)d_in[14];
  const float* W_zx  = (const float*)d_in[15];
  const float* b_z   = (const float*)d_in[16];
  const float* W_h   = (const float*)d_in[17];
  const float* W_x   = (const float*)d_in[18];
  const float* b     = (const float*)d_in[19];

  char* ws = (char*)d_ws;
  size_t off = 0;
  auto alloc = [&](size_t bytes) -> char* {
    char* p = ws + off;
    off += (bytes + 255) & ~(size_t)255;
    return p;
  };

  __bf16* hx    = (__bf16*)alloc((size_t)B_ROWS * K_CAT * 2);   // [h|x] bf16
  __bf16* Wc1   = (__bf16*)alloc((size_t)4 * D_H * K_CAT * 2);
  __bf16* WTsb  = (__bf16*)alloc((size_t)D_H * D_H * 2);
  __bf16* Wc3   = (__bf16*)alloc((size_t)D_H * K_CAT * 2);
  __bf16* sbf   = (__bf16*)alloc((size_t)B_ROWS * D_H * 2);
  float*  Tpart = (float*)alloc((size_t)B_ROWS * D_H * 4);
  float*  rbuf  = (float*)alloc((size_t)B_ROWS * D_H * 4);
  float*  zbuf  = (float*)alloc((size_t)B_ROWS * D_H * 4);

  {
    int total8 = B_ROWS * K_CAT / 8;
    pack_hx_kernel<<<(total8 + 255) / 256, 256, 0, stream>>>(h, x, hx, total8);
  }
  {
    PackW1Args pa;
    pa.Wh[0] = W_sh; pa.Wx[0] = W_sx;
    pa.Wh[1] = WTh;  pa.Wx[1] = WTx;
    pa.Wh[2] = W_rh; pa.Wx[2] = W_rx;
    pa.Wh[3] = W_zh; pa.Wx[3] = W_zx;
    pa.out = Wc1;
    int total8 = 4 * D_H * K_CAT / 8;
    pack_w1_kernel<<<(total8 + 255) / 256, 256, 0, stream>>>(pa, total8);
  }
  {
    int total8 = D_H * D_H / 8;
    cvt_bf16_kernel<<<(total8 + 255) / 256, 256, 0, stream>>>(WTs, WTsb, total8);
  }
  {
    int total8 = D_H * K_CAT / 8;
    pack_w3_kernel<<<(total8 + 255) / 256, 256, 0, stream>>>(W_h, W_x, Wc3, total8);
  }

  GemmParams g = {};
  g.delta = delta; g.Wst = W_st;
  g.bs = b_s; g.bT = b_T; g.br = b_r; g.bz = b_z;
  g.s_bf = sbf; g.Tpart = Tpart; g.rbuf = rbuf; g.zbuf = zbuf;
  g.b3 = b; g.hprev = h; g.out = (float*)d_out;

  // Stage 1: fused 4-gate GEMM  [B x 1536] @ [4096 x 1536]^T
  g.A = hx; g.W = Wc1; g.K = K_CAT;
  gemm_wmma_kernel<1><<<dim3((4 * D_H) / NT, B_ROWS / MT), 256, 0, stream>>>(g);

  // Stage 2: T_t = sigmoid(s @ WTs^T + Tpart)
  g.A = sbf; g.W = WTsb; g.K = D_H;
  gemm_wmma_kernel<2><<<dim3(D_H / NT, B_ROWS / MT), 256, 0, stream>>>(g);

  // u = r * h overwrites h-half of packed activations
  {
    int total8 = B_ROWS * D_H / 8;
    u_write_kernel<<<(total8 + 255) / 256, 256, 0, stream>>>(rbuf, h, hx, total8);
  }

  // Stage 3: h_t = (1-z)*(T*h) + z*tanh([u|x] @ [W_h|W_x]^T + b)
  g.A = hx; g.W = Wc3; g.K = K_CAT;
  gemm_wmma_kernel<3><<<dim3(D_H / NT, B_ROWS / MT), 256, 0, stream>>>(g);

  (void)in_sizes; (void)n_in; (void)out_size; (void)ws_size;
}
